// MoELayer_22600117912041
// MI455X (gfx1250) — compile-verified, matrix operand order checked by reading, not run on hardware
//
#include <hip/hip_runtime.h>

#define DIM 1024
#define HID 2048
#define NEXP 8
#define NTOK 8192            // 4 * 2048
#define INV_SQRT2 0.70710678118f

typedef __attribute__((ext_vector_type(16))) __bf16 v16bf;
typedef __attribute__((ext_vector_type(8)))  float  v8f;
typedef unsigned short us;   // bf16 storage

// ---------------- helpers ----------------

__device__ __forceinline__ us f2b(float f) {
    union { float f; unsigned u; } x; x.f = f;
    unsigned r = x.u + 0x7FFFu + ((x.u >> 16) & 1u);   // RNE
    return (us)(r >> 16);
}

__device__ __forceinline__ v16bf load_frag(const us* p) {
    // fragment layout (§7.12.2, 16-bit A/B): halves 0..7 = K[k0..k0+7],
    // halves 8..15 = K[k0+16..k0+23]; caller passes p = row + k0.
    union { v16bf v; uint4 q[2]; } f;
    f.q[0] = *(const uint4*)(p);
    f.q[1] = *(const uint4*)(p + 16);
    return f.v;
}

__device__ __forceinline__ v8f wmma_bf16(v16bf a, v16bf b, v8f c) {
    return __builtin_amdgcn_wmma_f32_16x16x32_bf16(false, a, false, b, (short)0, c,
                                                   false, false);
}

// CDNA5 async global->LDS copy (16B per lane), tracked by ASYNCcnt (§08).
// LDS operand = low 32 bits of the generic pointer (flat-LDS truncation, §10.2).
__device__ __forceinline__ void async_cp16(const us* g, us* l) {
    asm volatile("global_load_async_to_lds_b128 %0, %1, off"
                 :: "v"((unsigned)(uintptr_t)l),
                    "v"((unsigned long long)(uintptr_t)g)
                 : "memory");
}

__device__ __forceinline__ void wait_async0() {
    asm volatile("s_wait_asynccnt 0" ::: "memory");
}

// ---------------- kernel 1: gating + x->bf16 ----------------
// 256 threads = 8 waves, one token per wave.
__global__ void moe_gate(const float* __restrict__ x,
                         const float* __restrict__ gate_w,
                         const float* __restrict__ gate_b,
                         us* __restrict__ xb,
                         int* __restrict__ top2i, float* __restrict__ top2w,
                         float* __restrict__ usage, int* __restrict__ counts) {
    __shared__ float susage[NEXP];
    __shared__ int   scnt[NEXP];
    const int tid  = threadIdx.x;
    const int wave = tid >> 5, lane = tid & 31;
    if (tid < NEXP) { susage[tid] = 0.f; scnt[tid] = 0; }
    __syncthreads();

    const int token = blockIdx.x * 8 + wave;
    const float* xr = x + (size_t)token * DIM;
    unsigned* xbr   = (unsigned*)(xb + (size_t)token * DIM);

    float acc[NEXP];
#pragma unroll
    for (int e = 0; e < NEXP; ++e) acc[e] = 0.f;

#pragma unroll 4
    for (int it = 0; it < 16; ++it) {
        const int i2 = lane + it * 32;           // dword (pair) index
        const int d  = i2 * 2;
        const float2 xv = *(const float2*)(xr + d);
        const float* g0 = gate_w + (size_t)d * NEXP;
        const float* g1 = g0 + NEXP;
#pragma unroll
        for (int e = 0; e < NEXP; ++e) acc[e] += xv.x * g0[e] + xv.y * g1[e];
        xbr[i2] = (unsigned)f2b(xv.x) | ((unsigned)f2b(xv.y) << 16);
    }
    // butterfly reduce across wave32
#pragma unroll
    for (int off = 16; off; off >>= 1)
#pragma unroll
        for (int e = 0; e < NEXP; ++e) acc[e] += __shfl_xor(acc[e], off, 32);

    if (lane == 0) {
        float sc[NEXP], mx = -1e30f;
#pragma unroll
        for (int e = 0; e < NEXP; ++e) { sc[e] = acc[e] + gate_b[e]; mx = fmaxf(mx, sc[e]); }
        float sum = 0.f;
#pragma unroll
        for (int e = 0; e < NEXP; ++e) { sc[e] = __expf(sc[e] - mx); sum += sc[e]; }
        const float inv = 1.f / sum;
        int i1 = 0; float p1 = -1.f;
#pragma unroll
        for (int e = 0; e < NEXP; ++e) {
            sc[e] *= inv;
            atomicAdd(&susage[e], sc[e]);
            if (sc[e] > p1) { p1 = sc[e]; i1 = e; }
        }
        int i2 = (i1 == 0) ? 1 : 0; float p2 = -1.f;
#pragma unroll
        for (int e = 0; e < NEXP; ++e)
            if (e != i1 && sc[e] > p2) { p2 = sc[e]; i2 = e; }
        atomicAdd(&scnt[i1], 1);
        atomicAdd(&scnt[i2], 1);
        top2i[2 * token]     = i1; top2w[2 * token]     = p1;
        top2i[2 * token + 1] = i2; top2w[2 * token + 1] = p2;
    }
    __syncthreads();
    if (tid < NEXP) {
        atomicAdd(&usage[tid],  susage[tid]);
        atomicAdd(&counts[tid], scnt[tid]);
    }
}

// ---------------- kernel 2: fp32 [b][R][C] -> bf16 [b][C][R] (transpose+cvt) ----
__global__ void moe_tcvt(const float* __restrict__ in, us* __restrict__ out,
                         int R, int C) {
    __shared__ float tile[32][33];
    const int bx = blockIdx.x, by = blockIdx.y, bz = blockIdx.z;
    const int tx = threadIdx.x, ty = threadIdx.y;
    const size_t base = (size_t)bz * R * C;
#pragma unroll
    for (int i = 0; i < 4; ++i) {
        const int r = by * 32 + ty + i * 8;
        tile[ty + i * 8][tx] = in[base + (size_t)r * C + bx * 32 + tx];
    }
    __syncthreads();
#pragma unroll
    for (int i = 0; i < 4; ++i) {
        const int c = bx * 32 + ty + i * 8;
        out[base + (size_t)c * R + by * 32 + tx] = f2b(tile[tx][ty + i * 8]);
    }
}

// ---------------- kernel 3: exclusive prefix over 8 counts ----------------
__global__ void moe_offsets(const int* __restrict__ counts, int* __restrict__ offs) {
    if (threadIdx.x == 0) {
        int run = 0;
        for (int e = 0; e < NEXP; ++e) { offs[e] = run; run += counts[e]; }
    }
}

// ---------------- kernel 4: fill compact per-expert slot lists ----------------
__global__ void moe_fill(const int* __restrict__ top2i, const float* __restrict__ top2w,
                         const int* __restrict__ offs, int* __restrict__ poscnt,
                         int* __restrict__ slot_tok, float* __restrict__ slot_w) {
    const int t = blockIdx.x * blockDim.x + threadIdx.x;
    if (t >= NTOK) return;
#pragma unroll
    for (int j = 0; j < 2; ++j) {
        const int e = top2i[2 * t + j];
        const int pos = atomicAdd(&poscnt[e], 1);
        const int s = offs[e] + pos;
        slot_tok[s] = t;
        slot_w[s]   = top2w[2 * t + j] * INV_SQRT2;
    }
}

// ---------------- GEMM tiling constants ----------------
#define TM 128
#define TN 128
#define KC 64
#define SAS 72   // LDS row stride in halves (144B, 16B-aligned, conflict-skewed)

// ---------------- kernel 5: act = (Xe@W1+b1) * silu(Xe@W2+b2), bf16 out ------
// grid: (HID/TN, NTOK/TM, NEXP), block 256 (8 waves, wave tile 32x64)
// Double-buffered LDS fed by async global->LDS copies overlapped with WMMA.
__global__ void __launch_bounds__(256)
moe_gemm1(const us* __restrict__ xb,
          const us* __restrict__ wb1, const us* __restrict__ wb2,
          const float* __restrict__ b1, const float* __restrict__ b2,
          const int* __restrict__ slot_tok,
          const int* __restrict__ counts, const int* __restrict__ offs,
          us* __restrict__ act) {
    const int e     = blockIdx.z;
    const int count = counts[e];
    const int mBase = blockIdx.y * TM;
    if (mBase >= count) return;
    const int nBase = blockIdx.x * TN;
    const int offE  = offs[e];

    __shared__ int rowTok[TM];
    __shared__ __align__(16) us sA [2][TM * SAS];
    __shared__ __align__(16) us sB1[2][TN * SAS];
    __shared__ __align__(16) us sB2[2][TN * SAS];

    const int tid = threadIdx.x;
    if (tid < TM) {
        int s = mBase + tid; if (s >= count) s = count - 1;
        rowTok[tid] = slot_tok[offE + s];
    }
    __syncthreads();

    const int wave = tid >> 5, lane = tid & 31;
    const int wm = wave >> 1, wn = wave & 1;
    const int l15 = lane & 15, lHi = lane >> 4;

    const us* wb1e = wb1 + (size_t)e * HID * DIM + (size_t)nBase * DIM;
    const us* wb2e = wb2 + (size_t)e * HID * DIM + (size_t)nBase * DIM;

    // async-stage one K-chunk into LDS buffer `buf`
    auto stage = [&](int kb, int buf) {
#pragma unroll
        for (int i = tid; i < TM * 8; i += 256) {
            const int row = i >> 3, seg = i & 7;
            async_cp16(xb + (size_t)rowTok[row] * DIM + kb + seg * 8,
                       &sA[buf][row * SAS + seg * 8]);
        }
#pragma unroll
        for (int i = tid; i < TN * 8; i += 256) {
            const int row = i >> 3, seg = i & 7;
            async_cp16(wb1e + (size_t)row * DIM + kb + seg * 8,
                       &sB1[buf][row * SAS + seg * 8]);
            async_cp16(wb2e + (size_t)row * DIM + kb + seg * 8,
                       &sB2[buf][row * SAS + seg * 8]);
        }
    };

    v8f acc1[2][4], acc2[2][4];
#pragma unroll
    for (int i = 0; i < 2; ++i)
#pragma unroll
        for (int j = 0; j < 4; ++j) { acc1[i][j] = {}; acc2[i][j] = {}; }

    stage(0, 0);
    wait_async0();
    __syncthreads();

    for (int it = 0; it < DIM / KC; ++it) {
        const int cur = it & 1;
        if (it + 1 < DIM / KC) stage((it + 1) * KC, cur ^ 1);   // overlap with WMMA
#pragma unroll
        for (int kk = 0; kk < KC; kk += 32) {
            v16bf aF[2];
#pragma unroll
            for (int mt = 0; mt < 2; ++mt)
                aF[mt] = load_frag(&sA[cur][(wm * 32 + mt * 16 + l15) * SAS + kk + lHi * 8]);
#pragma unroll
            for (int nt = 0; nt < 4; ++nt) {
                const int bro = (wn * 64 + nt * 16 + l15) * SAS + kk + lHi * 8;
                v16bf bF = load_frag(&sB1[cur][bro]);
#pragma unroll
                for (int mt = 0; mt < 2; ++mt) acc1[mt][nt] = wmma_bf16(aF[mt], bF, acc1[mt][nt]);
                bF = load_frag(&sB2[cur][bro]);
#pragma unroll
                for (int mt = 0; mt < 2; ++mt) acc2[mt][nt] = wmma_bf16(aF[mt], bF, acc2[mt][nt]);
            }
        }
        wait_async0();
        __syncthreads();
    }

    // epilogue: bias + SiLU-gate, bf16 store to compact activation buffer
#pragma unroll
    for (int nt = 0; nt < 4; ++nt) {
        const int n = nBase + wn * 64 + nt * 16 + l15;
        const float bb1 = b1[(size_t)e * HID + n];
        const float bb2 = b2[(size_t)e * HID + n];
#pragma unroll
        for (int mt = 0; mt < 2; ++mt)
#pragma unroll
            for (int r = 0; r < 8; ++r) {
                const int srow = mBase + wm * 32 + mt * 16 + lHi * 8 + r;
                if (srow < count) {
                    const float h = acc1[mt][nt][r] + bb1;
                    const float g = acc2[mt][nt][r] + bb2;
                    const float a = h * (g / (1.f + __expf(-g)));
                    act[(size_t)(offE + srow) * HID + n] = f2b(a);
                }
            }
    }
}

// ---------------- kernel 6: out += gate * (act@W3 + b3), atomic scatter -------
// grid: (DIM/TN, NTOK/TM, NEXP), block 256
__global__ void __launch_bounds__(256)
moe_gemm2(const us* __restrict__ act, const us* __restrict__ wb3,
          const float* __restrict__ b3,
          const int* __restrict__ slot_tok, const float* __restrict__ slot_w,
          const int* __restrict__ counts, const int* __restrict__ offs,
          float* __restrict__ out) {
    const int e     = blockIdx.z;
    const int count = counts[e];
    const int mBase = blockIdx.y * TM;
    if (mBase >= count) return;
    const int nBase = blockIdx.x * TN;
    const int offE  = offs[e];

    __shared__ __align__(16) us sA[2][TM * SAS];
    __shared__ __align__(16) us sB[2][TN * SAS];

    const int tid = threadIdx.x;
    const int wave = tid >> 5, lane = tid & 31;
    const int wm = wave >> 1, wn = wave & 1;
    const int l15 = lane & 15, lHi = lane >> 4;

    const us* wb3e = wb3 + (size_t)e * DIM * HID + (size_t)nBase * HID;

    auto stage = [&](int kb, int buf) {
#pragma unroll
        for (int i = tid; i < TM * 8; i += 256) {
            const int row = i >> 3, seg = i & 7;
            int rr = mBase + row; if (rr >= count) rr = count - 1;
            async_cp16(act + (size_t)(offE + rr) * HID + kb + seg * 8,
                       &sA[buf][row * SAS + seg * 8]);
        }
#pragma unroll
        for (int i = tid; i < TN * 8; i += 256) {
            const int row = i >> 3, seg = i & 7;
            async_cp16(wb3e + (size_t)row * HID + kb + seg * 8,
                       &sB[buf][row * SAS + seg * 8]);
        }
    };

    v8f acc[2][4];
#pragma unroll
    for (int i = 0; i < 2; ++i)
#pragma unroll
        for (int j = 0; j < 4; ++j) acc[i][j] = {};

    stage(0, 0);
    wait_async0();
    __syncthreads();

    for (int it = 0; it < HID / KC; ++it) {
        const int cur = it & 1;
        if (it + 1 < HID / KC) stage((it + 1) * KC, cur ^ 1);
#pragma unroll
        for (int kk = 0; kk < KC; kk += 32) {
            v16bf aF[2];
#pragma unroll
            for (int mt = 0; mt < 2; ++mt)
                aF[mt] = load_frag(&sA[cur][(wm * 32 + mt * 16 + l15) * SAS + kk + lHi * 8]);
#pragma unroll
            for (int nt = 0; nt < 4; ++nt) {
                v16bf bF = load_frag(&sB[cur][(wn * 64 + nt * 16 + l15) * SAS + kk + lHi * 8]);
#pragma unroll
                for (int mt = 0; mt < 2; ++mt) acc[mt][nt] = wmma_bf16(aF[mt], bF, acc[mt][nt]);
            }
        }
        wait_async0();
        __syncthreads();
    }

#pragma unroll
    for (int nt = 0; nt < 4; ++nt) {
        const int n = nBase + wn * 64 + nt * 16 + l15;
        const float bb3 = b3[(size_t)e * DIM + n];
#pragma unroll
        for (int mt = 0; mt < 2; ++mt)
#pragma unroll
            for (int r = 0; r < 8; ++r) {
                const int srow = mBase + wm * 32 + mt * 16 + lHi * 8 + r;
                if (srow < count) {
                    const int s = offE + srow;
                    const int token = slot_tok[s];
                    const float w = slot_w[s];          // gate_score * INV_SQRT2
                    atomicAdd(&out[(size_t)token * DIM + n], w * (acc[mt][nt][r] + bb3));
                }
            }
    }
}

// ---------------- kernel 7: load-balance entropy loss ----------------
__global__ void moe_lb(const float* __restrict__ usage, float* __restrict__ out) {
    if (threadIdx.x == 0) {
        const float inv = 1.f / (float)NTOK;
        float lb = 0.f;
        for (int e = 0; e < NEXP; ++e) {
            const float u = usage[e] * inv;
            lb -= u * __logf(u + 1e-9f);
        }
        out[(size_t)NTOK * DIM] = lb;   // appended after the [b,s,d] tensor
    }
}

// ---------------- host launcher ----------------
extern "C" void kernel_launch(void* const* d_in, const int* in_sizes, int n_in,
                              void* d_out, int out_size, void* d_ws, size_t ws_size,
                              hipStream_t stream) {
    const float* x      = (const float*)d_in[0];
    const float* gate_w = (const float*)d_in[1];
    const float* gate_b = (const float*)d_in[2];
    const float* w1     = (const float*)d_in[3];
    const float* b1     = (const float*)d_in[4];
    const float* w2     = (const float*)d_in[5];
    const float* b2     = (const float*)d_in[6];
    const float* w3     = (const float*)d_in[7];
    const float* b3     = (const float*)d_in[8];
    float* out = (float*)d_out;

    // workspace layout
    char* W = (char*)d_ws;
    float* usage   = (float*)(W);            // 8 f32
    int*   counts  = (int*)(W + 32);         // 8 i32
    int*   poscnt  = (int*)(W + 64);         // 8 i32
    int*   offs    = (int*)(W + 96);         // 8 i32
    int*   top2i   = (int*)(W + 256);                     // 16384 i32
    float* top2w   = (float*)(W + 256 + 1 * 65536);       // 16384 f32
    int*   slot_tok= (int*)(W + 256 + 2 * 65536);         // 16384 i32
    float* slot_w  = (float*)(W + 256 + 3 * 65536);       // 16384 f32
    us* xb  = (us*)(W + 262400);                          // 8192x1024 bf16 (16MB)
    us* wb1 = xb  + (size_t)NTOK * DIM;                   // 8x2048x1024 (32MB) [e][h][d]
    us* wb2 = wb1 + (size_t)NEXP * DIM * HID;             // 32MB
    us* wb3 = wb2 + (size_t)NEXP * DIM * HID;             // 8x1024x2048 (32MB) [e][d][h]
    us* actb= wb3 + (size_t)NEXP * DIM * HID;             // 16384x2048 bf16 (64MB)

    hipMemsetAsync(d_out, 0, (size_t)out_size * sizeof(float), stream);
    hipMemsetAsync(d_ws, 0, 128, stream);   // usage/counts/poscnt/offs

    moe_gate<<<NTOK / 8, 256, 0, stream>>>(x, gate_w, gate_b, xb, top2i, top2w,
                                           usage, counts);

    dim3 tb(32, 8);
    moe_tcvt<<<dim3(HID / 32, DIM / 32, NEXP), tb, 0, stream>>>(w1, wb1, DIM, HID);
    moe_tcvt<<<dim3(HID / 32, DIM / 32, NEXP), tb, 0, stream>>>(w2, wb2, DIM, HID);
    moe_tcvt<<<dim3(DIM / 32, HID / 32, NEXP), tb, 0, stream>>>(w3, wb3, HID, DIM);

    moe_offsets<<<1, 32, 0, stream>>>(counts, offs);
    moe_fill<<<NTOK / 256, 256, 0, stream>>>(top2i, top2w, offs, poscnt,
                                             slot_tok, slot_w);

    moe_gemm1<<<dim3(HID / TN, NTOK / TM, NEXP), 256, 0, stream>>>(
        xb, wb1, wb2, b1, b2, slot_tok, counts, offs, actb);

    moe_gemm2<<<dim3(DIM / TN, NTOK / TM, NEXP), 256, 0, stream>>>(
        actb, wb3, b3, slot_tok, slot_w, counts, offs, out);

    moe_lb<<<1, 32, 0, stream>>>(usage, out);
}
